// SingleHeadAttention_74929999446738
// MI455X (gfx1250) — matrix operand order repaired and checked by reading, not compiled, not run on hardware
//
#include <hip/hip_runtime.h>

#define DEVINL __device__ __forceinline__

typedef __attribute__((ext_vector_type(16))) _Float16 v16h;
typedef __attribute__((ext_vector_type(8)))  _Float16 v8h;
typedef __attribute__((ext_vector_type(8)))  float    v8f;
typedef __attribute__((ext_vector_type(4)))  float    v4f;
typedef __attribute__((ext_vector_type(4)))  unsigned int v4u;
typedef __attribute__((ext_vector_type(8)))  int      v8i;
typedef __attribute__((ext_vector_type(4)))  int      v4i;

constexpr int B = 4, S = 4096, H = 1024, D = 128;
constexpr float SCALE = 0.08838834764831845f; // 1/sqrt(128)

// padded LDS row strides (halfs)
constexpr int KSTR = 136;   // 128 + 8  (pad 4 dwords every 64 dwords)
constexpr int VSTR = 40;    // 32  + 8  (pad 4 dwords every 16 dwords)
constexpr int PSTR = 40;

// ---- workspace layout (in _Float16 elements) ----
constexpr size_t OFF_WTK = 0;
constexpr size_t OFF_WTQ = OFF_WTK + (size_t)D * H;
constexpr size_t OFF_WTV = OFF_WTQ + (size_t)D * H;
constexpr size_t OFF_QH  = OFF_WTV + (size_t)D * H;        // q  [B*S][D] f16
constexpr size_t OFF_KH  = OFF_QH  + (size_t)B * S * D;    // k  [B*S][D] f16
constexpr size_t OFF_VT  = OFF_KH  + (size_t)B * S * D;    // vT [B][D][S] f16

#if defined(__has_builtin)
#  if __has_builtin(__builtin_amdgcn_tensor_load_to_lds) && \
      __has_builtin(__builtin_amdgcn_s_wait_tensorcnt)
#    define USE_TDM 1
#  endif
#endif
#ifndef USE_TDM
#  define USE_TDM 0
#endif

DEVINL v8f wmma_f16(v16h a, v16h b, v8f c) {
  return __builtin_amdgcn_wmma_f32_16x16x32_f16(false, a, false, b, (short)0, c,
                                                false, false);
}

DEVINL v16h cat16(v8h lo, v8h hi) {
  v16h r;
#pragma unroll
  for (int i = 0; i < 8; ++i) { r[i] = lo[i]; r[i + 8] = hi[i]; }
  return r;
}

DEVINL v16h cvt16(v4f a0, v4f a1, v4f a2, v4f a3) {
  v16h r;
#pragma unroll
  for (int i = 0; i < 4; ++i) {
    r[i]      = (_Float16)a0[i];
    r[i + 4]  = (_Float16)a1[i];
    r[i + 8]  = (_Float16)a2[i];
    r[i + 12] = (_Float16)a3[i];
  }
  return r;
}

DEVINL float rowmax16(float v) {
#pragma unroll
  for (int m = 1; m < 16; m <<= 1) v = fmaxf(v, __shfl_xor(v, m, 32));
  return v;
}
DEVINL float rowsum16(float v) {
#pragma unroll
  for (int m = 1; m < 16; m <<= 1) v += __shfl_xor(v, m, 32);
  return v;
}

#if USE_TDM
// 2D TDM load: tile (tile_d1 lines x tile_d0 f16 elements) from global to LDS,
// with LDS padding pad_amount(code) dwords every pad_interval(code) dwords.
DEVINL void tdm_load_2d(unsigned lds_addr, const void* gsrc,
                        unsigned tile_d0, unsigned tile_d1,
                        unsigned tensor_d0, unsigned tensor_d1,
                        unsigned stride0_elts,
                        unsigned pad_interval_code, unsigned pad_amount_code) {
  unsigned long long ga = (unsigned long long)(size_t)gsrc;
  v4u g0;
  g0[0] = 1u;                                        // count=1, user descriptor
  g0[1] = lds_addr;                                  // LDS byte address
  g0[2] = (unsigned)(ga & 0xffffffffu);              // global_addr[31:0]
  g0[3] = (unsigned)((ga >> 32) & 0x1ffffffu)        // global_addr[56:32]
        | (2u << 30);                                // type = 2 ("image")
  v8i g1;
  g1[0] = (int)((1u << 16)                           // data_size = 2 bytes
        | (1u << 20)                                 // pad_enable
        | (pad_interval_code << 22)
        | (pad_amount_code << 25));
  g1[1] = (int)((tensor_d0 & 0xffffu) << 16);        // tensor_dim0[15:0]
  g1[2] = (int)((tensor_d0 >> 16) | ((tensor_d1 & 0xffffu) << 16));
  g1[3] = (int)((tensor_d1 >> 16) | (tile_d0 << 16));
  g1[4] = (int)(tile_d1 & 0xffffu);                  // tile_dim1 (tile_dim2=0)
  g1[5] = (int)stride0_elts;                         // tensor_dim0_stride[31:0]
  g1[6] = 0;
  g1[7] = 0;
  // force descriptor dwords into SGPRs
#pragma unroll
  for (int i = 0; i < 4; ++i)
    g0[i] = (unsigned)__builtin_amdgcn_readfirstlane((int)g0[i]);
#pragma unroll
  for (int i = 0; i < 8; ++i)
    g1[i] = __builtin_amdgcn_readfirstlane(g1[i]);
  v4i z4 = {0, 0, 0, 0};
#if defined(__clang_major__) && (__clang_major__ >= 23)
  v8i z8 = {0, 0, 0, 0, 0, 0, 0, 0};
  __builtin_amdgcn_tensor_load_to_lds(g0, g1, z4, z4, z8, 0);
#else
  __builtin_amdgcn_tensor_load_to_lds(g0, g1, z4, z4, 0);
#endif
}
#endif

// ---------------------------------------------------------------------------
// Kernel 1: transpose + convert the three weight matrices to f16 [D][H]
// ---------------------------------------------------------------------------
__global__ void prep_weights(const float* __restrict__ Wk,
                             const float* __restrict__ Wq,
                             const float* __restrict__ Wv,
                             _Float16* __restrict__ ws) {
  int idx = blockIdx.x * blockDim.x + threadIdx.x;
  if (idx >= 3 * H * D) return;
  int w = idx / (H * D);
  int r = idx % (H * D);
  int n = r / H;  // output row (d)
  int k = r % H;  // h
  const float* src = (w == 0) ? Wk : (w == 1) ? Wq : Wv;
  _Float16* dst = ws + ((w == 0) ? OFF_WTK : (w == 1) ? OFF_WTQ : OFF_WTV);
  dst[(size_t)n * H + k] = (_Float16)src[(size_t)k * D + n];
}

// ---------------------------------------------------------------------------
// Kernel 2: q = x@Wq + bq, k = x@Wk + bk  (f16 out, row-major [B*S][D])
// ---------------------------------------------------------------------------
__global__ void __launch_bounds__(256) proj_qk(const float* __restrict__ x,
                                               const float* __restrict__ bk,
                                               const float* __restrict__ bq,
                                               _Float16* __restrict__ ws) {
  const int lane = threadIdx.x & 31, wave = threadIdx.x >> 5;
  const int ln = lane & 15, lh = lane >> 4;
  int id   = blockIdx.x * 8 + wave;     // 0..16383
  int mat  = id >> 13;                  // 0 = q, 1 = k
  int rem  = id & 8191;
  int row0 = (rem >> 3) << 4;
  int col0 = (rem & 7) << 4;
  const _Float16* wt   = ws + (mat ? OFF_WTK : OFF_WTQ);
  const float*    bias = mat ? bk : bq;
  _Float16*       outp = ws + (mat ? OFF_KH : OFF_QH);

  float bval = bias[col0 + ln];
  v8f acc;
#pragma unroll
  for (int r = 0; r < 8; ++r) acc[r] = bval;

  const float*    xrow = x  + (size_t)(row0 + ln) * H;
  const _Float16* wrow = wt + (size_t)(col0 + ln) * H;

  for (int h0 = 0; h0 < H; h0 += 32) {
    const float* ap = xrow + h0 + lh * 8;
    v16h a = cvt16(*(const v4f*)(ap),      *(const v4f*)(ap + 4),
                   *(const v4f*)(ap + 16), *(const v4f*)(ap + 20));
    const _Float16* bp = wrow + h0 + lh * 8;
    v16h b = cat16(*(const v8h*)(bp), *(const v8h*)(bp + 16));
    acc = wmma_f16(a, b, acc);
  }
#pragma unroll
  for (int r = 0; r < 8; ++r)
    outp[(size_t)(row0 + r + lh * 8) * D + col0 + ln] = (_Float16)acc[r];
}

// ---------------------------------------------------------------------------
// Kernel 3: vT = (x@Wv + bv)^T computed directly as WvT @ xT -> [B][D][S] f16
// ---------------------------------------------------------------------------
__global__ void __launch_bounds__(256) proj_vt(const float* __restrict__ x,
                                               const float* __restrict__ bv,
                                               _Float16* __restrict__ ws) {
  const int lane = threadIdx.x & 31, wave = threadIdx.x >> 5;
  const int ln = lane & 15, lh = lane >> 4;
  int id  = blockIdx.x * 8 + wave;      // 0..8191
  int b   = id >> 11;
  int rem = id & 2047;
  int d0  = (rem >> 8) << 4;
  int s0  = (rem & 255) << 4;

  const _Float16* wrow = ws + OFF_WTV + (size_t)(d0 + ln) * H;
  const float*    xrow = x + ((size_t)b * S + s0 + ln) * H;

  v8f acc;
#pragma unroll
  for (int r = 0; r < 8; ++r) acc[r] = bv[d0 + r + lh * 8];

  for (int h0 = 0; h0 < H; h0 += 32) {
    const _Float16* ap = wrow + h0 + lh * 16;
    v16h a = cat16(*(const v8h*)(ap), *(const v8h*)(ap + 8));
    const float* bp = xrow + h0 + lh * 16;
    v16h bb = cvt16(*(const v4f*)(bp),     *(const v4f*)(bp + 4),
                    *(const v4f*)(bp + 8), *(const v4f*)(bp + 12));
    acc = wmma_f16(a, bb, acc);
  }
  _Float16* vt = ws + OFF_VT + (size_t)b * D * S;
#pragma unroll
  for (int r = 0; r < 8; ++r)
    vt[(size_t)(d0 + r + lh * 8) * S + s0 + ln] = (_Float16)acc[r];
}

// ---------------------------------------------------------------------------
// Kernel 4: causal flash attention.
// Block = 8 waves = 128 consecutive queries of one batch. K/V 32-key blocks
// staged in double-buffered LDS by the Tensor Data Mover (TDM) and shared by
// all waves; per-wave online softmax; all GEMMs via v_wmma_f32_16x16x32_f16.
// ---------------------------------------------------------------------------
__global__ void __launch_bounds__(256) flash_attn(const _Float16* __restrict__ wsh,
                                                  float* __restrict__ out) {
  __shared__ __align__(128) _Float16 Kls[2][32 * KSTR];   // 17408 B
  __shared__ __align__(128) _Float16 Vls[2][128 * VSTR];  // 20480 B
  __shared__ __align__(128) _Float16 Pls[8][16 * PSTR];   // 10240 B

  const int lane = threadIdx.x & 31, wave = threadIdx.x >> 5;
  const int ln = lane & 15, lh = lane >> 4;
  const int b      = blockIdx.x >> 5;          // 32 blocks per batch
  const int qbase0 = (blockIdx.x & 31) << 7;   // 128 queries per block
  const int qbase  = qbase0 + wave * 16;

  const _Float16* qh  = wsh + OFF_QH;
  const _Float16* khb = wsh + OFF_KH + (size_t)b * S * D;
  const _Float16* vtb = wsh + OFF_VT + (size_t)b * D * S;

  // Q fragments: lane = query row, 4 d-chunks of 32
  const _Float16* qrow = qh + ((size_t)b * S + qbase + ln) * D;
  v16h Qf[4];
#pragma unroll
  for (int dc = 0; dc < 4; ++dc) {
    const _Float16* p = qrow + dc * 32 + lh * 16;
    Qf[dc] = cat16(*(const v8h*)(p), *(const v8h*)(p + 8));
  }

  v8f O[8];
#pragma unroll
  for (int t = 0; t < 8; ++t)
#pragma unroll
    for (int r = 0; r < 8; ++r) O[t][r] = 0.0f;

  float m8[8], l8[8];
#pragma unroll
  for (int r = 0; r < 8; ++r) { m8[r] = -3.0e38f; l8[r] = 0.0f; }

  const int my_nkb = (qbase + 47) >> 5;
  const int nkbmax = (qbase0 + 159) >> 5;   // trip count uniform across block

  // ---- prologue: stage key-block 0 into buffer 0 ----
#if USE_TDM
  if (wave == 0) {
    tdm_load_2d((unsigned)(size_t)&Kls[0][0], khb, 128, 32, 128, 32, 128, 5, 3);
    tdm_load_2d((unsigned)(size_t)&Vls[0][0], vtb, 32, 128, 32, 128, S, 3, 3);
  }
#else
  for (int c = threadIdx.x; c < 512; c += 256) {
    int kr = c >> 4, kc = (c & 15) * 8;
    *(v8h*)(&Kls[0][0] + kr * KSTR + kc) =
        *(const v8h*)(khb + (size_t)kr * D + kc);
    int vr = c >> 2, vc = (c & 3) * 8;
    *(v8h*)(&Vls[0][0] + vr * VSTR + vc) =
        *(const v8h*)(vtb + (size_t)vr * S + vc);
  }
#endif

  for (int kbi = 0; kbi < nkbmax; ++kbi) {
    const int cur = kbi & 1;
    // ---- stage next key block into the other buffer ----
#if USE_TDM
    if (wave == 0) {
      if (kbi + 1 < nkbmax) {
        const int kb2 = (kbi + 1) << 5;
        tdm_load_2d((unsigned)(size_t)&Kls[cur ^ 1][0],
                    khb + (size_t)kb2 * D, 128, 32, 128, 32, 128, 5, 3);
        tdm_load_2d((unsigned)(size_t)&Vls[cur ^ 1][0],
                    vtb + kb2, 32, 128, 32, 128, S, 3, 3);
        __builtin_amdgcn_s_wait_tensorcnt(2);  // current block resident
      } else {
        __builtin_amdgcn_s_wait_tensorcnt(0);
      }
    }
#else
    if (kbi + 1 < nkbmax) {
      const int kb2 = (kbi + 1) << 5;
      for (int c = threadIdx.x; c < 512; c += 256) {
        int kr = c >> 4, kc = (c & 15) * 8;
        *(v8h*)(&Kls[cur ^ 1][0] + kr * KSTR + kc) =
            *(const v8h*)(khb + (size_t)(kb2 + kr) * D + kc);
        int vr = c >> 2, vc = (c & 3) * 8;
        *(v8h*)(&Vls[cur ^ 1][0] + vr * VSTR + vc) =
            *(const v8h*)(vtb + (size_t)vr * S + kb2 + vc);
      }
    }
#endif
    __syncthreads();   // current buffer visible to all waves

    if (kbi < my_nkb) {
      const int kb = kbi << 5;
      // ---- scores from LDS K ----
      const _Float16* Kb0 = &Kls[cur][0] + (size_t)ln * KSTR;
      const _Float16* Kb1 = Kb0 + 16 * KSTR;
      v8f S0, S1;
#pragma unroll
      for (int r = 0; r < 8; ++r) { S0[r] = 0.0f; S1[r] = 0.0f; }
#pragma unroll
      for (int dc = 0; dc < 4; ++dc) {
        const _Float16* p0 = Kb0 + dc * 32 + lh * 16;
        const _Float16* p1 = Kb1 + dc * 32 + lh * 16;
        v16h B0 = cat16(*(const v8h*)(p0), *(const v8h*)(p0 + 8));
        v16h B1 = cat16(*(const v8h*)(p1), *(const v8h*)(p1 + 8));
        S0 = wmma_f16(Qf[dc], B0, S0);
        S1 = wmma_f16(Qf[dc], B1, S1);
      }
      // ---- online softmax ----
      float p0v[8], p1v[8];
#pragma unroll
      for (int r = 0; r < 8; ++r) {
        const int row = qbase + r + lh * 8;
        const int c0  = kb + ln;
        float s0 = (c0      <= row) ? S0[r] * SCALE : -3.0e38f;
        float s1 = (c0 + 16 <= row) ? S1[r] * SCALE : -3.0e38f;
        float mx   = rowmax16(fmaxf(s0, s1));
        float mnew = fmaxf(m8[r], mx);
        float corr = __expf(m8[r] - mnew);
        float p0   = __expf(s0 - mnew);
        float p1   = __expf(s1 - mnew);
        float ps   = rowsum16(p0 + p1);
        l8[r] = l8[r] * corr + ps;
        m8[r] = mnew;
#pragma unroll
        for (int t = 0; t < 8; ++t) O[t][r] *= corr;
        p0v[r] = p0; p1v[r] = p1;
      }
      // ---- C-layout -> A-layout for P via per-wave LDS round trip ----
      _Float16* pbuf = Pls[wave];
#pragma unroll
      for (int r = 0; r < 8; ++r) {
        const int rr = r + lh * 8;
        pbuf[rr * PSTR + ln]      = (_Float16)p0v[r];
        pbuf[rr * PSTR + ln + 16] = (_Float16)p1v[r];
      }
      asm volatile("s_wait_dscnt 0" ::: "memory");
      v8h plo = *(const v8h*)(pbuf + ln * PSTR + lh * 16);
      v8h phi = *(const v8h*)(pbuf + ln * PSTR + lh * 16 + 8);
      v16h Pf = cat16(plo, phi);
      // ---- O += P @ V from LDS vT ----
      const _Float16* Vb = &Vls[cur][0];
#pragma unroll
      for (int t = 0; t < 8; ++t) {
        const _Float16* vp = Vb + (size_t)(t * 16 + ln) * VSTR + lh * 16;
        v16h Vf = cat16(*(const v8h*)(vp), *(const v8h*)(vp + 8));
        O[t] = wmma_f16(Pf, Vf, O[t]);
      }
    }
    __syncthreads();   // everyone done with current buffer before overwrite
  }

  // ---- epilogue: normalize and store fp32 ----
  float* orow = out + ((size_t)b * S + qbase) * D;
#pragma unroll
  for (int r = 0; r < 8; ++r) {
    const float inv = 1.0f / l8[r];
#pragma unroll
    for (int t = 0; t < 8; ++t)
      orow[(size_t)(r + lh * 8) * D + t * 16 + ln] = O[t][r] * inv;
  }
}

// ---------------------------------------------------------------------------
extern "C" void kernel_launch(void* const* d_in, const int* in_sizes, int n_in,
                              void* d_out, int out_size, void* d_ws, size_t ws_size,
                              hipStream_t stream) {
  const float* x  = (const float*)d_in[0];
  const float* Wk = (const float*)d_in[1];
  const float* bk = (const float*)d_in[2];
  const float* Wq = (const float*)d_in[3];
  const float* bq = (const float*)d_in[4];
  const float* Wv = (const float*)d_in[5];
  const float* bv = (const float*)d_in[6];
  float*     out = (float*)d_out;
  _Float16*  ws  = (_Float16*)d_ws;

  prep_weights<<<(3 * H * D + 255) / 256, 256, 0, stream>>>(Wk, Wq, Wv, ws);
  proj_qk<<<2048, 256, 0, stream>>>(x, bk, bq, ws);   // 16384 wave-tiles
  proj_vt<<<1024, 256, 0, stream>>>(x, bv, ws);       // 8192 wave-tiles
  flash_attn<<<128, 256, 0, stream>>>(ws, out);       // 1024 query-tile waves
}
